// Message_Passing_Core_24781961297936
// MI455X (gfx1250) — matrix operand order repaired and checked by reading, not compiled, not run on hardware
//
#include <hip/hip_runtime.h>
#include <hip/hip_bf16.h>

typedef __attribute__((ext_vector_type(16))) __bf16 v16bf;
typedef __attribute__((ext_vector_type(8)))  __bf16 v8bf;
typedef __attribute__((ext_vector_type(8)))  float  v8f;
typedef __attribute__((ext_vector_type(4)))  float  v4f;

#define NBATCH 4
#define DF     64          // all feature dims are 64
#define NLIT   8192
#define NCLS   16384
#define SLOPE  0.01f
#define MROWS  128         // output rows per block (8 waves x 16)

__device__ __forceinline__ float leaky(float x) { return x >= 0.0f ? x : SLOPE * x; }

// Async DMA: copy 16 bytes global -> LDS (per lane), tracked by ASYNCcnt.
__device__ __forceinline__ void async_b128(unsigned lds_addr, const void* gaddr) {
  asm volatile("global_load_async_to_lds_b128 %0, %1, off"
               :: "v"(lds_addr), "v"(gaddr)
               : "memory");
}
__device__ __forceinline__ void wait_async0() {
  asm volatile("s_wait_asynccnt 0x0" ::: "memory");
}

// ---------------------------------------------------------------------------
// prep: LMbf[b][f][l] = bf16( W_Lmsg @ L_t + b_Lmsg )   (feature-major, K=NL)
// ---------------------------------------------------------------------------
__global__ void prep_lm_kernel(const float* __restrict__ Lt,
                               const float* __restrict__ W,   // [64][64]
                               const float* __restrict__ bias,
                               __bf16* __restrict__ LMbf) {
  __shared__ __align__(16) float ldsW[DF * DF];
  const int tid = threadIdx.x;
  for (int idx = tid; idx < DF * DF; idx += 256) ldsW[idx] = W[idx];
  __syncthreads();

  const int b = blockIdx.y;
  const int l = blockIdx.x * 256 + tid;

  float x[DF];
#pragma unroll
  for (int i = 0; i < DF; ++i)
    x[i] = Lt[(size_t)(b * DF + i) * NLIT + l];

#pragma unroll 1
  for (int f = 0; f < DF; ++f) {
    float acc = bias[f];
#pragma unroll
    for (int i = 0; i < DF; ++i) acc += ldsW[f * DF + i] * x[i];
    LMbf[(size_t)(b * DF + f) * NLIT + l] = (__bf16)acc;
  }
}

// ---------------------------------------------------------------------------
// Core WMMA tile loop shared by both big GEMMs (8 waves, 128 rows x 64 feats).
// D[m, n] = sum_k Adj[mbase+m][k] * Bmat[b][n][k]   (bf16 x bf16 -> f32)
// B K-chunks (64x32 bf16 = 4 KB) are DMA'd global->LDS with ASYNC loads,
// double-buffered: DMA of chunk i+1 overlaps WMMA on chunk i.
// ---------------------------------------------------------------------------
__device__ __forceinline__ void wmma_tile_loop(const float* __restrict__ Adj,
                                               const __bf16* __restrict__ Bmat,
                                               int K, int b, int mbase,
                                               __bf16* ldsB /*2 x 2048*/,
                                               float* ldsD /*128 x 65*/) {
  const int tid  = threadIdx.x;
  const int lane = tid & 31;
  const int wv   = tid >> 5;   // 0..7
  const int hs   = lane >> 4;  // lane-half
  const int lr   = lane & 15;

  const float* arow = Adj + (size_t)(mbase + wv * 16 + lr) * K;

  v8f acc[4];
#pragma unroll
  for (int nt = 0; nt < 4; ++nt)
#pragma unroll
    for (int v = 0; v < 8; ++v) acc[nt][v] = 0.0f;

  // staging: 256 threads x 16B = one 64x32 bf16 chunk per iteration
  const int stg_n = tid >> 2;      // feature row 0..63
  const int stg_q = tid & 3;       // 8-element quarter of the K-chunk
  const __bf16* stg_src = Bmat + (size_t)(b * DF + stg_n) * K + stg_q * 8;
  const unsigned stg_dst =
      (unsigned)(size_t)ldsB + (unsigned)(stg_n * 32 + stg_q * 8) * 2u;

  // prologue: DMA chunk 0 into buffer 0
  async_b128(stg_dst, stg_src);
  wait_async0();
  __syncthreads();

#pragma unroll 1
  for (int kk = 0; kk < K; kk += 32) {
    const int cur = (kk >> 5) & 1;
    if (kk + 32 < K)  // DMA next chunk into the other buffer (overlaps WMMA)
      async_b128(stg_dst + (unsigned)(1 - cur) * 4096u, stg_src + kk + 32);

    // A fragment: elems 0..7 <- K = kk+hs*8.., elems 8..15 <- K = kk+16+hs*8..
    // Non-temporal: each adjacency element is read exactly once.
    const float* ap = arow + kk + hs * 8;
    v4f f0 = __builtin_nontemporal_load((const v4f*)ap);
    v4f f1 = __builtin_nontemporal_load((const v4f*)(ap + 4));
    v4f f2 = __builtin_nontemporal_load((const v4f*)(ap + 16));
    v4f f3 = __builtin_nontemporal_load((const v4f*)(ap + 20));
    if (kk + 32 < K) __builtin_prefetch(ap + 32, 0, 0);
    v16bf afrag;
#pragma unroll
    for (int e = 0; e < 4; ++e) {
      afrag[e]      = (__bf16)f0[e];
      afrag[4 + e]  = (__bf16)f1[e];
      afrag[8 + e]  = (__bf16)f2[e];
      afrag[12 + e] = (__bf16)f3[e];
    }

    const __bf16* bbuf = ldsB + cur * 2048;
#pragma unroll
    for (int nt = 0; nt < 4; ++nt) {
      const __bf16* bp = bbuf + (nt * 16 + lr) * 32 + hs * 8;
      v8bf b0 = *(const v8bf*)bp;
      v8bf b1 = *(const v8bf*)(bp + 16);
      v16bf bfrag = __builtin_shufflevector(b0, b1, 0, 1, 2, 3, 4, 5, 6, 7,
                                            8, 9, 10, 11, 12, 13, 14, 15);
      acc[nt] = __builtin_amdgcn_wmma_f32_16x16x32_bf16(
          false, afrag, false, bfrag, (short)0, acc[nt], false, false);
    }

    if (kk + 32 < K) wait_async0();  // our DMA into next buffer is done
    __syncthreads();                 // everyone done reading cur / writing next
  }

  // spill D tile: lanes 0-15: M=v, N=lane; lanes 16-31: M=v+8, N=lane-16
#pragma unroll
  for (int nt = 0; nt < 4; ++nt)
#pragma unroll
    for (int v = 0; v < 8; ++v)
      ldsD[(wv * 16 + hs * 8 + v) * 65 + nt * 16 + lr] = acc[nt][v];
}

// ---------------------------------------------------------------------------
// GEMM1: M_l tile + fused clause update + CM (bf16) production
// ---------------------------------------------------------------------------
__global__ void gemm1_kernel(const float* __restrict__ Aadj,   // [NC][NL]
                             const __bf16* __restrict__ LMbf,  // [b][64][NL]
                             const float* __restrict__ Ct,     // [b][64][NC]
                             const float* __restrict__ WCu,    // [64][128]
                             const float* __restrict__ bCu,
                             const float* __restrict__ WCm,    // [64][64]
                             const float* __restrict__ bCm,
                             float* __restrict__ outC,         // [b][64][NC]
                             __bf16* __restrict__ CMbf) {      // [b][64][NC]
  __shared__ __align__(16) __bf16 ldsB[2 * DF * 32];
  __shared__ __align__(16) float  ldsD[MROWS * 65];
  __shared__ __align__(16) float  ldsCT[DF * MROWS];
  __shared__ __align__(16) float  ldsW1[DF * 128];
  __shared__ __align__(16) float  ldsW2[DF * DF];

  const int tid   = threadIdx.x;
  const int b     = blockIdx.y;
  const int cbase = blockIdx.x * MROWS;

  wmma_tile_loop(Aadj, LMbf, NLIT, b, cbase, ldsB, ldsD);

  for (int idx = tid; idx < DF * MROWS; idx += 256) {
    int i = idx >> 7, c = idx & 127;
    ldsCT[idx] = Ct[(size_t)(b * DF + i) * NCLS + cbase + c];
  }
  for (int idx = tid; idx < DF * 128; idx += 256) ldsW1[idx] = WCu[idx];
  for (int idx = tid; idx < DF * DF; idx += 256)  ldsW2[idx] = WCm[idx];
  __syncthreads();

  const int c  = tid & 127;
  const int fg = tid >> 7;   // wave-uniform feature group (0/1)
  float ctnew[32];

#pragma unroll 1
  for (int j = 0; j < 32; ++j) {
    const int f = fg * 32 + j;
    float a = bCu[f];
    const float* w = &ldsW1[f * 128];
#pragma unroll
    for (int i = 0; i < DF; ++i)
      a += w[i] * ldsCT[i * 128 + c] + w[64 + i] * ldsD[c * 65 + i];
    a = leaky(a);
    ctnew[j] = a;
    outC[(size_t)(b * DF + f) * NCLS + cbase + c] = a;
  }
  __syncthreads();               // all reads of M_l tile done
#pragma unroll
  for (int j = 0; j < 32; ++j) ldsD[c * 65 + fg * 32 + j] = ctnew[j];
  __syncthreads();

#pragma unroll 1
  for (int j = 0; j < 32; ++j) {
    const int f = fg * 32 + j;
    float a = bCm[f];
    const float* w = &ldsW2[f * DF];
#pragma unroll
    for (int i = 0; i < DF; ++i) a += w[i] * ldsD[c * 65 + i];
    CMbf[(size_t)(b * DF + f) * NCLS + cbase + c] = (__bf16)a;
  }
}

// ---------------------------------------------------------------------------
// GEMM2: M_c tile + fused literal update (with flip)
// ---------------------------------------------------------------------------
__global__ void gemm2_kernel(const float* __restrict__ At,     // [NL][NC]
                             const __bf16* __restrict__ CMbf,  // [b][64][NC]
                             const float* __restrict__ Lt,     // [b][64][NL]
                             const float* __restrict__ WLu,    // [64][192]
                             const float* __restrict__ bLu,
                             float* __restrict__ outL) {       // [b][64][NL]
  __shared__ __align__(16) __bf16 ldsB[2 * DF * 32];
  __shared__ __align__(16) float  ldsD[MROWS * 65];
  __shared__ __align__(16) float  ldsLT[DF * MROWS];
  __shared__ __align__(16) float  ldsLF[DF * MROWS];
  __shared__ __align__(16) float  ldsW[DF * 192];

  const int tid   = threadIdx.x;
  const int b     = blockIdx.y;
  const int lbase = blockIdx.x * MROWS;
  const int fbase = (lbase + NLIT / 2) % NLIT;   // flip block (contiguous)

  wmma_tile_loop(At, CMbf, NCLS, b, lbase, ldsB, ldsD);

  for (int idx = tid; idx < DF * MROWS; idx += 256) {
    int i = idx >> 7, l = idx & 127;
    ldsLT[idx] = Lt[(size_t)(b * DF + i) * NLIT + lbase + l];
    ldsLF[idx] = Lt[(size_t)(b * DF + i) * NLIT + fbase + l];
  }
  for (int idx = tid; idx < DF * 192; idx += 256) ldsW[idx] = WLu[idx];
  __syncthreads();

  const int l  = tid & 127;
  const int fg = tid >> 7;

#pragma unroll 1
  for (int j = 0; j < 32; ++j) {
    const int f = fg * 32 + j;
    float a = bLu[f];
    const float* w = &ldsW[f * 192];
#pragma unroll
    for (int i = 0; i < DF; ++i)
      a += w[i] * ldsLT[i * 128 + l] + w[64 + i] * ldsD[l * 65 + i] +
           w[128 + i] * ldsLF[i * 128 + l];
    outL[(size_t)(b * DF + f) * NLIT + lbase + l] = leaky(a);
  }
}

// ---------------------------------------------------------------------------
// row sums (for the global/U update)
// ---------------------------------------------------------------------------
__global__ void rowsum_kernel(const float* __restrict__ src,
                              float* __restrict__ dst, int ncols) {
  __shared__ float red[256];
  const int row = blockIdx.x;
  const float* p = src + (size_t)row * ncols;
  float s = 0.0f;
  for (int i = threadIdx.x; i < ncols; i += 256) s += p[i];
  red[threadIdx.x] = s;
  __syncthreads();
  for (int off = 128; off > 0; off >>= 1) {
    if (threadIdx.x < off) red[threadIdx.x] += red[threadIdx.x + off];
    __syncthreads();
  }
  if (threadIdx.x == 0) dst[row] = red[0];
}

// ---------------------------------------------------------------------------
// U update: leaky(W_Uu @ [Lsum; Csum; U_t] + b_Uu)
// ---------------------------------------------------------------------------
__global__ void ufinal_kernel(const float* __restrict__ Lsum,
                              const float* __restrict__ Csum,
                              const float* __restrict__ Ut,
                              const float* __restrict__ WUu,   // [64][192]
                              const float* __restrict__ bUu,
                              float* __restrict__ outU) {
  const int b = blockIdx.x;
  const int f = threadIdx.x;
  float a = bUu[f];
  const float* w = &WUu[f * 192];
#pragma unroll
  for (int i = 0; i < DF; ++i) a += w[i] * Lsum[b * DF + i];
#pragma unroll
  for (int i = 0; i < DF; ++i) a += w[64 + i] * Csum[b * DF + i];
#pragma unroll
  for (int i = 0; i < DF; ++i) a += w[128 + i] * Ut[b * DF + i];
  outU[b * DF + f] = leaky(a);
}

// ---------------------------------------------------------------------------
extern "C" void kernel_launch(void* const* d_in, const int* in_sizes, int n_in,
                              void* d_out, int out_size, void* d_ws, size_t ws_size,
                              hipStream_t stream) {
  (void)in_sizes; (void)n_in; (void)out_size; (void)ws_size;

  const float* Lt    = (const float*)d_in[0];
  const float* Ct    = (const float*)d_in[1];
  const float* Ut    = (const float*)d_in[2];
  const float* Aadj  = (const float*)d_in[3];
  const float* At    = (const float*)d_in[4];
  const float* WLmsg = (const float*)d_in[5];
  const float* bLmsg = (const float*)d_in[6];
  const float* WCmsg = (const float*)d_in[7];
  const float* bCmsg = (const float*)d_in[8];
  const float* WLu   = (const float*)d_in[9];
  const float* bLu   = (const float*)d_in[10];
  const float* WCu   = (const float*)d_in[11];
  const float* bCu   = (const float*)d_in[12];
  const float* WUu   = (const float*)d_in[13];
  const float* bUu   = (const float*)d_in[14];

  float* outL = (float*)d_out;                              // [4][64][NL]
  float* outC = outL + (size_t)NBATCH * DF * NLIT;          // [4][64][NC]
  float* outU = outC + (size_t)NBATCH * DF * NCLS;          // [4][64]

  char* ws = (char*)d_ws;
  __bf16* LMbf = (__bf16*)ws;                                         // 4 MB
  __bf16* CMbf = (__bf16*)(ws + (size_t)NBATCH * DF * NLIT * 2);      // 8 MB
  float*  Lsum = (float*)(ws + (size_t)NBATCH * DF * NLIT * 2
                             + (size_t)NBATCH * DF * NCLS * 2);
  float*  Csum = Lsum + NBATCH * DF;

  prep_lm_kernel<<<dim3(NLIT / 256, NBATCH), 256, 0, stream>>>(Lt, WLmsg, bLmsg, LMbf);

  gemm1_kernel<<<dim3(NCLS / MROWS, NBATCH), 256, 0, stream>>>(
      Aadj, LMbf, Ct, WCu, bCu, WCmsg, bCmsg, outC, CMbf);

  gemm2_kernel<<<dim3(NLIT / MROWS, NBATCH), 256, 0, stream>>>(
      At, CMbf, Lt, WLu, bLu, outL);

  rowsum_kernel<<<NBATCH * DF, 256, 0, stream>>>(outL, Lsum, NLIT);
  rowsum_kernel<<<NBATCH * DF, 256, 0, stream>>>(outC, Csum, NCLS);

  ufinal_kernel<<<NBATCH, DF, 0, stream>>>(Lsum, Csum, Ut, WUu, bUu, outU);
}
